// CausalSelfAttention_14645838479365
// MI455X (gfx1250) — matrix operand order repaired and checked by reading, not compiled, not run on hardware
//
#include <hip/hip_runtime.h>
#include <hip/hip_bf16.h>

#define DEV __device__ __forceinline__

typedef __attribute__((ext_vector_type(16))) __bf16 v16bf;
typedef __attribute__((ext_vector_type(8)))  float  v8f;
typedef unsigned int u32x4 __attribute__((ext_vector_type(4)));
typedef int          i32x4 __attribute__((ext_vector_type(4)));
typedef int          i32x8 __attribute__((ext_vector_type(8)));
typedef __attribute__((address_space(3))) unsigned short lds_us;

union FragAB {
  unsigned int u[8];
  v16bf v;
};

DEV unsigned short f2bf(float f) {
  unsigned int u = __float_as_uint(f);
  u += 0x7FFFu + ((u >> 16) & 1u);   // round-to-nearest-even
  return (unsigned short)(u >> 16);
}

// A-matrix fragment (16x32 bf16): lane row = lane%16, half = lane/16.
// VGPR g element pair at K = 16*half + 2*g + {0,1}
DEV void loadA(const unsigned short* __restrict__ p, long row, int ld, int k0,
               int hh, unsigned int* u) {
  const unsigned short* rp = p + row * (long)ld + k0 + hh * 16;
#pragma unroll
  for (int g = 0; g < 8; ++g) {
    int kk = g * 2;
    u[g] = *(const unsigned int*)(rp + kk);
  }
}

// B-matrix fragment (32x16 bf16): lane col = lane%16, half = lane/16.
// VGPR g element pair at K = 16*half + 2*g + {0,1}
DEV void loadB(const unsigned short* __restrict__ p, long row, int ld, int k0,
               int hh, unsigned int* u) {
  const unsigned short* rp = p + row * (long)ld + k0 + hh * 16;
#pragma unroll
  for (int g = 0; g < 8; ++g) u[g] = *(const unsigned int*)(rp + 2 * g);
}

DEV v8f wmma_bf16(const FragAB& a, const FragAB& b, v8f c) {
  return __builtin_amdgcn_wmma_f32_16x16x32_bf16(false, a.v, false, b.v,
                                                 (short)0, c, false, false);
}

// TDM: load a 64(row) x 32(col) bf16 tile of a (N x K) row-major tensor to LDS.
// D# per CDNA5 ISA ch.8: group0 = count/lds_addr/global_addr/type,
// group1 = data_size, tensor dims, tile dims, stride.
DEV void tdm_issue_tile(unsigned ldsOff, unsigned long long gaddr, int K, int N) {
  u32x4 g0;
  g0[0] = 1u;                                       // count=1, user mode
  g0[1] = ldsOff;                                   // lds_addr (bytes)
  g0[2] = (unsigned)gaddr;                          // global_addr[31:0]
  g0[3] = (unsigned)((gaddr >> 32) & 0x01FFFFFFull) // global_addr[56:32]
          | 0x80000000u;                            // type=2 ("image")
  i32x8 g1;
  g1[0] = 0x00010000;                               // data_size=1 -> 2 bytes
  g1[1] = (K & 0xFFFF) << 16;                       // tensor_dim0[15:0]
  g1[2] = ((K >> 16) & 0xFFFF) | ((N & 0xFFFF) << 16);   // dim0 hi | dim1 lo
  g1[3] = ((N >> 16) & 0xFFFF) | (32 << 16);        // dim1 hi | tile_dim0=32
  g1[4] = 64;                                       // tile_dim1=64
  g1[5] = K;                                        // tensor_dim0_stride lo32
  g1[6] = 0;
  g1[7] = 0;
  i32x4 z4 = {0, 0, 0, 0};
  i32x8 z8 = {0, 0, 0, 0, 0, 0, 0, 0};
  __builtin_amdgcn_tensor_load_to_lds(g0, g1, z4, z4, z8, 0);
}

// ---------------------------------------------------------------- cvt kernels
__global__ void cvt_bf16_kernel(const float* __restrict__ in,
                                unsigned short* __restrict__ out, int n) {
  int i = blockIdx.x * 256 + threadIdx.x;
  if (i < n) out[i] = f2bf(in[i]);
}

// w: K x N row-major (f32)  ->  wt: N x K row-major (bf16)
__global__ void transpose_cvt_kernel(const float* __restrict__ w,
                                     unsigned short* __restrict__ wt,
                                     int K, int N) {
  int i = blockIdx.x * 256 + threadIdx.x;
  if (i < K * N) {
    int n = i % N, k = i / N;
    wt[(long)n * K + k] = f2bf(w[i]);
  }
}

// -------------------------------------------------------------------- GEMM
// C[M,N] = A[M,K](bf16) * Bt[N,K](bf16, pre-transposed) + bias[N], fp32 out.
// Block tile 128(M) x 64(N): 8 waves x (16x64). The shared 64x32 B-tile per
// K-step is DMA'd to LDS by the TDM (double-buffered, issued by wave 0),
// giving 8x reuse out of LDS instead of 8 redundant L2 streams.
__global__ __launch_bounds__(256) void gemm_bf16_kernel(
    const unsigned short* __restrict__ A, const unsigned short* __restrict__ Bt,
    const float* __restrict__ bias, float* __restrict__ C,
    int M, int N, int K) {
  __shared__ unsigned short ldsB[2][64 * 32];
  int wave = threadIdx.x >> 5, lane = threadIdx.x & 31;
  int r = lane & 15, hh = lane >> 4;
  int nblks = N >> 6;
  int mblk = blockIdx.x / nblks, nblk = blockIdx.x % nblks;
  long m0 = (long)mblk * 128 + wave * 16;
  long n0 = (long)nblk * 64;

  unsigned ldsOff0 = (unsigned)(unsigned long long)(lds_us*)&ldsB[0][0];
  unsigned ldsOff1 = (unsigned)(unsigned long long)(lds_us*)&ldsB[1][0];
  unsigned long long bBase = (unsigned long long)Bt + (n0 * (long)K) * 2ll;

  int nk = K >> 5;
  if (wave == 0)  // prologue: tile 0 -> buffer 0
    tdm_issue_tile(ldsOff0, bBase, K, N);

  v8f acc[4] = {};
  for (int j = 0; j < nk; ++j) {
    if (wave == 0) {
      if (j + 1 < nk) {
        tdm_issue_tile((j & 1) ? ldsOff0 : ldsOff1,
                       bBase + (unsigned long long)(j + 1) * 64ull, K, N);
        __builtin_amdgcn_s_wait_tensorcnt(1);   // tile j complete
      } else {
        __builtin_amdgcn_s_wait_tensorcnt(0);
      }
    }
    __syncthreads();                            // B tile j visible to all waves

    int k0 = j * 32;
    FragAB a;
    loadA(A, m0 + r, K, k0, hh, a.u);
    if (k0 + 64 < K)
      __builtin_prefetch(A + (m0 + r) * (long)K + k0 + 64, 0, 1);
    const unsigned short* btile = &ldsB[j & 1][0];
#pragma unroll
    for (int ch = 0; ch < 4; ++ch) {
      FragAB b;
      const unsigned short* lp = btile + (ch * 16 + r) * 32 + hh * 16;
#pragma unroll
      for (int g = 0; g < 8; ++g) b.u[g] = *(const unsigned int*)(lp + 2 * g);
      acc[ch] = wmma_bf16(a, b, acc[ch]);
    }
    __syncthreads();                            // release buffer j&1
  }

#pragma unroll
  for (int ch = 0; ch < 4; ++ch) {
    long n = n0 + ch * 16 + r;
    float bv = bias[n];
#pragma unroll
    for (int v = 0; v < 8; ++v) {
      long m = m0 + v + 8 * hh;
      C[m * N + n] = acc[ch][v] + bv;
    }
  }
}

// ---------------------------------------------------------------- RoPE + pack
// qkv: (B*T, 6144) fp32.  ->  Qb,Kb: (bh, t, 128) bf16 (q pre-scaled by 1/sqrt(128))
//                             Vt: (bh, 128, T) bf16 (transposed for B-fragments)
__global__ void rope_pack_kernel(const float* __restrict__ qkv,
                                 unsigned short* __restrict__ Qb,
                                 unsigned short* __restrict__ Kb,
                                 unsigned short* __restrict__ Vt) {
  int idx = blockIdx.x * 256 + threadIdx.x;      // (b,h,t,i) i<64
  if (idx >= 4 * 16 * 2048 * 64) return;
  int i = idx & 63;
  int rem = idx >> 6;
  int t = rem & 2047;
  rem >>= 11;
  int h = rem & 15, b = rem >> 4;

  const float* row = qkv + (long)(b * 2048 + t) * 6144;
  int d0 = h * 128 + i, d1 = d0 + 64;
  float q1 = row[d0],        q2 = row[d1];
  float k1 = row[2048 + d0], k2 = row[2048 + d1];
  float v1 = row[4096 + d0], v2 = row[4096 + d1];

  float invf = __powf(10000.0f, -(2.0f * (float)i) / 128.0f);
  float ang = (float)t * invf;
  float s, c;
  __sincosf(ang, &s, &c);

  const float scale = 0.08838834764831845f;  // 1/sqrt(128)
  float qo1 = (q1 * c - q2 * s) * scale, qo2 = (q2 * c + q1 * s) * scale;
  float ko1 = k1 * c - k2 * s,           ko2 = k2 * c + k1 * s;

  long bh = b * 16 + h;
  unsigned short* Qrow = Qb + (bh * 2048 + t) * 128;
  unsigned short* Krow = Kb + (bh * 2048 + t) * 128;
  Qrow[i] = f2bf(qo1); Qrow[i + 64] = f2bf(qo2);
  Krow[i] = f2bf(ko1); Krow[i + 64] = f2bf(ko2);
  Vt[(bh * 128 + i) * 2048 + t]      = f2bf(v1);
  Vt[(bh * 128 + i + 64) * 2048 + t] = f2bf(v2);
}

// ------------------------------------------------------------ flash attention
// Block = 8 waves; each wave owns 16 q rows of one (b,h). Online softmax over
// 32-wide K tiles. ctx written bf16 into (b, t, h*128+d) layout for out-proj.
__global__ __launch_bounds__(256) void attn_kernel(
    const unsigned short* __restrict__ Qb, const unsigned short* __restrict__ Kb,
    const unsigned short* __restrict__ Vt, unsigned short* __restrict__ Ctx) {
  __shared__ float lds[8][16 * 32];
  int wave = threadIdx.x >> 5, lane = threadIdx.x & 31;
  int r = lane & 15, hh = lane >> 4;
  int qblk = blockIdx.x & 15;
  int bh = blockIdx.x >> 4;
  int q0 = qblk * 128 + wave * 16;

  const unsigned short* Qrow = Qb + (long)bh * 2048 * 128;
  const unsigned short* Krow = Kb + (long)bh * 2048 * 128;
  const unsigned short* Vrow = Vt + (long)bh * 128 * 2048;

  FragAB qf[4];
#pragma unroll
  for (int c4 = 0; c4 < 4; ++c4) loadA(Qrow, q0 + r, 128, 32 * c4, hh, qf[c4].u);

  float mrun[8], lrun[8];
  v8f acc[8] = {};
#pragma unroll
  for (int v = 0; v < 8; ++v) { mrun[v] = -3.0e38f; lrun[v] = 0.0f; }

  int ntile = (q0 + 16 + 31) >> 5;
  for (int j = 0; j < ntile; ++j) {
    int kb = j * 32;
    // ---- S = Q K^T for 32 k-columns (two 16-col halves)
    v8f s0 = {}, s1 = {};
#pragma unroll
    for (int c4 = 0; c4 < 4; ++c4) {
      FragAB kf0, kf1;
      loadB(Krow, kb + r, 128, 32 * c4, hh, kf0.u);
      loadB(Krow, kb + 16 + r, 128, 32 * c4, hh, kf1.u);
      s0 = wmma_bf16(qf[c4], kf0, s0);
      s1 = wmma_bf16(qf[c4], kf1, s1);
    }
    // ---- causal mask + online softmax (per row = vgpr + 8*hh)
    int col0 = kb + r, col1 = col0 + 16;
#pragma unroll
    for (int v = 0; v < 8; ++v) {
      int rowm = q0 + v + 8 * hh;
      float a0 = (col0 > rowm) ? -3.0e38f : s0[v];
      float a1 = (col1 > rowm) ? -3.0e38f : s1[v];
      float mx = fmaxf(a0, a1);
#pragma unroll
      for (int off = 8; off > 0; off >>= 1) mx = fmaxf(mx, __shfl_xor(mx, off, 32));
      float mn = fmaxf(mrun[v], mx);
      float sc = __expf(mrun[v] - mn);
      float p0 = __expf(a0 - mn);
      float p1 = __expf(a1 - mn);
      float rs = p0 + p1;
#pragma unroll
      for (int off = 8; off > 0; off >>= 1) rs += __shfl_xor(rs, off, 32);
      lrun[v] = lrun[v] * sc + rs;
      mrun[v] = mn;
#pragma unroll
      for (int n = 0; n < 8; ++n) acc[n][v] *= sc;
      int rowl = v + 8 * hh;
      lds[wave][rowl * 32 + r] = p0;          // C-layout -> LDS
      lds[wave][rowl * 32 + 16 + r] = p1;
    }
    asm volatile("s_wait_dscnt 0" ::: "memory");
    __builtin_amdgcn_wave_barrier();
    // ---- reload P as A-fragment (bf16)
    FragAB pf;
#pragma unroll
    for (int g = 0; g < 8; ++g) {
      int kk = hh * 16 + g * 2;
      float f0 = lds[wave][r * 32 + kk];
      float f1 = lds[wave][r * 32 + kk + 1];
      pf.u[g] = (unsigned int)f2bf(f0) | ((unsigned int)f2bf(f1) << 16);
    }
    // ---- ctx += P * V (V from transposed layout: contiguous k-pairs)
#pragma unroll
    for (int n = 0; n < 8; ++n) {
      FragAB vf;
      loadB(Vrow, n * 16 + r, 2048, kb, hh, vf.u);
      acc[n] = wmma_bf16(pf, vf, acc[n]);
    }
  }
  // ---- normalize and store ctx (bf16) into (b, t, h*128+d)
  int b = bh >> 4, h = bh & 15;
#pragma unroll
  for (int v = 0; v < 8; ++v) {
    float inv = 1.0f / lrun[v];
    int t = q0 + v + 8 * hh;
    unsigned short* crow = Ctx + (long)(b * 2048 + t) * 2048 + h * 128;
#pragma unroll
    for (int n = 0; n < 8; ++n) crow[n * 16 + r] = f2bf(acc[n][v] * inv);
  }
}

// ------------------------------------------------------------------- launch
extern "C" void kernel_launch(void* const* d_in, const int* in_sizes, int n_in,
                              void* d_out, int out_size, void* d_ws, size_t ws_size,
                              hipStream_t stream) {
  (void)in_sizes; (void)n_in; (void)out_size; (void)ws_size;
  const float* x     = (const float*)d_in[0];
  const float* w_qkv = (const float*)d_in[1];
  const float* b_qkv = (const float*)d_in[2];
  const float* w_out = (const float*)d_in[3];
  const float* b_out = (const float*)d_in[4];
  float* out = (float*)d_out;

  char* ws = (char*)d_ws;
  size_t off = 0;
  auto take = [&](size_t bytes) {
    char* p = ws + off;
    off += (bytes + 255) & ~(size_t)255;
    return p;
  };
  unsigned short* xbf   = (unsigned short*)take((size_t)8192 * 2048 * 2);
  unsigned short* wqkvT = (unsigned short*)take((size_t)6144 * 2048 * 2);
  unsigned short* woutT = (unsigned short*)take((size_t)2048 * 2048 * 2);
  float*          qkv   = (float*)take((size_t)8192 * 6144 * 4);
  unsigned short* Qbf   = (unsigned short*)take((size_t)64 * 2048 * 128 * 2);
  unsigned short* Kbf   = (unsigned short*)take((size_t)64 * 2048 * 128 * 2);
  unsigned short* Vtr   = (unsigned short*)take((size_t)64 * 128 * 2048 * 2);
  unsigned short* ctx   = (unsigned short*)take((size_t)8192 * 2048 * 2);

  cvt_bf16_kernel<<<65536, 256, 0, stream>>>(x, xbf, 8192 * 2048);
  transpose_cvt_kernel<<<49152, 256, 0, stream>>>(w_qkv, wqkvT, 2048, 6144);
  transpose_cvt_kernel<<<16384, 256, 0, stream>>>(w_out, woutT, 2048, 2048);
  // QKV projection: (8192 x 2048) * (2048 x 6144) -> fp32
  gemm_bf16_kernel<<<6144, 256, 0, stream>>>(xbf, wqkvT, b_qkv, qkv,
                                             8192, 6144, 2048);
  rope_pack_kernel<<<32768, 256, 0, stream>>>(qkv, Qbf, Kbf, Vtr);
  attn_kernel<<<1024, 256, 0, stream>>>(Qbf, Kbf, Vtr, ctx);
  // Output projection: (8192 x 2048) * (2048 x 2048) -> d_out
  gemm_bf16_kernel<<<2048, 256, 0, stream>>>(ctx, woutT, b_out, out,
                                             8192, 2048, 2048);
}
